// SlotAttentionWorkspace_790273983026
// MI455X (gfx1250) — compile-verified, matrix operand order checked by reading, not run on hardware
//
#include <hip/hip_runtime.h>
#include <hip/hip_bf16.h>
#include <math.h>

// ---------------------------------------------------------------------------
// Slot Attention fused pipeline for gfx1250 (MI455X), bf16 WMMA everywhere.
// B=64, M=4096, DIN=D=256, N=16, ITERS=3
// ---------------------------------------------------------------------------

#define B_   64
#define M_   4096
#define D_   256
#define N_   16
#define ITERS_ 3
#define SCALE_ 0.0625f      // 256^-0.5
#define LN_EPS_ 1e-5f

typedef __attribute__((ext_vector_type(16))) __bf16 v16bf;
typedef __attribute__((ext_vector_type(8)))  float  v8f;
typedef __attribute__((ext_vector_type(4)))  unsigned u32x4;
typedef __attribute__((ext_vector_type(8)))  unsigned u32x8;

// ---- WMMA helpers ---------------------------------------------------------

__device__ __forceinline__ v8f wmma_bf16(v16bf a, v16bf b, v8f c) {
  // D = A(16x32 bf16) * B(32x16 bf16) + C(16x16 f32)
  return __builtin_amdgcn_wmma_f32_16x16x32_bf16(
      /*neg_a=*/false, a, /*neg_b=*/false, b,
      /*c_mod=*/(short)0, c, /*reuse_a=*/false, /*reuse_b=*/false);
}

// A-fragment (16 rows x 32 K) from bf16 row-major storage.
// lane holds row = lane%16; element j -> K = off+j, element j+8 -> K = 16+off+j,
// off = (lane>=16) ? 8 : 0.  (ISA 7.12.2 16-bit A layout)
__device__ __forceinline__ v16bf frag_bf16(const __bf16* base, int stride, int lane) {
  int r   = lane & 15;
  int off = (lane & 16) ? 8 : 0;
  const __bf16* p = base + (long)r * stride + off;
  v16bf f;
#pragma unroll
  for (int j = 0; j < 8; ++j) { f[j] = p[j]; f[j + 8] = p[16 + j]; }
  return f;
}

// Same gather pattern but converting f32 -> bf16 on the fly (works on LDS ptr).
__device__ __forceinline__ v16bf frag_f32(const float* base, int stride, int lane) {
  int r   = lane & 15;
  int off = (lane & 16) ? 8 : 0;
  const float* p = base + (long)r * stride + off;
  v16bf f;
#pragma unroll
  for (int j = 0; j < 8; ++j) { f[j] = (__bf16)p[j]; f[j + 8] = (__bf16)p[16 + j]; }
  return f;
}

// ---- Tensor Data Mover: issue a 2D f32 tile load (rows x cols) into LDS ---
// D# per CDNA5 ISA (group0 + group1; groups 2/3 zeroed,
// tile_dim2/3 == 0 -> unused). Caller must be a single wave; follow with
// s_wait_tensorcnt(0) + workgroup barrier before other waves read the tile.
__device__ __forceinline__ void tdm_load_tile_f32(unsigned lds_addr,
                                                  unsigned long long gaddr,
                                                  unsigned tile_x, unsigned tile_y,
                                                  unsigned row_stride_elems) {
  u32x4 g0;
  g0[0] = 1u;                                    // count=1, user descriptor
  g0[1] = lds_addr;                              // LDS byte address
  g0[2] = (unsigned)(gaddr & 0xFFFFFFFFull);     // global_addr[31:0]
  g0[3] = (unsigned)((gaddr >> 32) & 0x1FFFFFFu) // global_addr[56:32]
          | (2u << 30);                          // type = 2 ("image")
  u32x8 g1;
  g1[0] = (2u << 16);                            // workgroup_mask=0, data_size=2 (4B)
  g1[1] = (tile_x & 0xFFFFu) << 16;              // tensor_dim0[15:0]  (== tile_x)
  g1[2] = (tile_x >> 16) | ((tile_y & 0xFFFFu) << 16);   // dim0 hi | tensor_dim1 lo
  g1[3] = (tile_y >> 16) | (tile_x << 16);       // tensor_dim1 hi | tile_dim0
  g1[4] = tile_y;                                // tile_dim1 | (tile_dim2=0)<<16
  g1[5] = row_stride_elems;                      // tensor_dim0_stride[31:0]
  g1[6] = 0u;                                    // stride hi | tensor_dim1_stride lo
  g1[7] = 0u;
  u32x4 gz = {0u, 0u, 0u, 0u};
  asm volatile("tensor_load_to_lds %0, %1, %2, %3"
               :: "s"(g0), "s"(g1), "s"(gz), "s"(gz)
               : "memory");
}

// ---- generic GEMM:  C[rows][cols] = A(f32)[rows][K] * Bt(bf16)[cols][K]^T --
// Each wave owns CT adjacent 16x16 output tiles (one A frag feeds CT WMMAs).
// A tile (16x32 f32) is staged into LDS by the Tensor Data Mover (wave 0).
// grid.x = rows/16, grid.y = cols/(8*CT*16), block = 256 (8 waves)
template <int CT>
__global__ void gemm_bf16t_kernel(const float* __restrict__ A, int lda,
                                  const __bf16* __restrict__ Bt, int ldb,
                                  const float* __restrict__ bias,
                                  float* __restrict__ Cf,
                                  __bf16* __restrict__ Cbf, int ldc,
                                  int K) {
  __shared__ float Atile[16 * 32];
  const int tid  = threadIdx.x;
  const int wave = tid >> 5;
  const int lane = tid & 31;
  const int row0 = blockIdx.x * 16;
  const int col0 = (blockIdx.y * 8 + wave) * (CT * 16);

  const unsigned atile_lds = (unsigned)(unsigned long long)(void*)Atile;
  const unsigned long long abase =
      (unsigned long long)(A + (long)row0 * lda);

  v8f acc[CT];
#pragma unroll
  for (int t = 0; t < CT; ++t) acc[t] = (v8f){};

  for (int k0 = 0; k0 < K; k0 += 32) {
    __syncthreads();                 // previous tile fully consumed
    if (wave == 0) {
      tdm_load_tile_f32(atile_lds, abase + (unsigned)k0 * 4ull,
                        /*tile_x=*/32u, /*tile_y=*/16u,
                        /*row_stride_elems=*/(unsigned)lda);
      __builtin_amdgcn_s_wait_tensorcnt(0);
    }
    __syncthreads();                 // publish LDS tile to all waves
    v16bf a = frag_f32(Atile, 32, lane);
#pragma unroll
    for (int t = 0; t < CT; ++t) {
      v16bf b = frag_bf16(Bt + (long)(col0 + t * 16) * ldb + k0, ldb, lane);
      acc[t] = wmma_bf16(a, b, acc[t]);
    }
  }

#pragma unroll
  for (int t = 0; t < CT; ++t) {
    const int col = col0 + t * 16 + (lane & 15);
    const float bv = bias ? bias[col] : 0.0f;
#pragma unroll
    for (int i = 0; i < 8; ++i) {
      int row = row0 + i + ((lane & 16) ? 8 : 0);
      float v = acc[t][i] + bv;
      if (Cf)  Cf[(long)row * ldc + col] = v;
      if (Cbf) Cbf[(long)row * ldc + col] = (__bf16)v;
    }
  }
}

// ---- attention logits:  logits[b][n][m] = SCALE * sum_d q[b,n,d]*k[b,m,d] --
// grid.x = B, grid.y = M/128, block = 256
__global__ void attn_logits_kernel(const __bf16* __restrict__ qbf,   // [B*16][256]
                                   const __bf16* __restrict__ kv,    // [B*4096][512] (k at col 0)
                                   float* __restrict__ logits) {     // [B][16][4096]
  const int b    = blockIdx.x;
  const int wave = threadIdx.x >> 5;
  const int lane = threadIdx.x & 31;
  const int m0   = (blockIdx.y * 8 + wave) * 16;
  const __bf16* qb = qbf + (long)b * 16 * 256;
  const __bf16* kb = kv + (long)b * 4096 * 512;

  v8f acc = {};
#pragma unroll
  for (int k0 = 0; k0 < 256; k0 += 32) {
    v16bf a = frag_bf16(qb + k0, 256, lane);
    v16bf bm = frag_bf16(kb + (long)m0 * 512 + k0, 512, lane);
    acc = wmma_bf16(a, bm, acc);
  }

  const int m = m0 + (lane & 15);
  float* lb = logits + (long)b * 16 * 4096;
#pragma unroll
  for (int i = 0; i < 8; ++i) {
    int n = i + ((lane & 16) ? 8 : 0);
    lb[(long)n * 4096 + m] = acc[i] * SCALE_;
  }
}

// ---- softmax over M=4096 per (b,n) row; writes normalized bf16 ------------
// grid = B*16, block = 256
__global__ void softmax_rows_kernel(const float* __restrict__ logits,
                                    __bf16* __restrict__ attn_bf) {
  const long row = blockIdx.x;
  const float* src = logits + row * 4096;
  __bf16* dst = attn_bf + row * 4096;
  __shared__ float red[8];
  const int tid = threadIdx.x;

  float m = -3.402823466e+38f;
  for (int i = tid; i < 4096; i += 256) m = fmaxf(m, src[i]);
#pragma unroll
  for (int o = 16; o > 0; o >>= 1) m = fmaxf(m, __shfl_xor(m, o, 32));
  if ((tid & 31) == 0) red[tid >> 5] = m;
  __syncthreads();
  float bm = red[0];
#pragma unroll
  for (int w = 1; w < 8; ++w) bm = fmaxf(bm, red[w]);
  __syncthreads();

  float s = 0.0f;
  for (int i = tid; i < 4096; i += 256) s += expf(src[i] - bm);
#pragma unroll
  for (int o = 16; o > 0; o >>= 1) s += __shfl_xor(s, o, 32);
  if ((tid & 31) == 0) red[tid >> 5] = s;
  __syncthreads();
  float bs = 0.0f;
#pragma unroll
  for (int w = 0; w < 8; ++w) bs += red[w];
  const float inv = 1.0f / bs;

  for (int i = tid; i < 4096; i += 256)
    dst[i] = (__bf16)(expf(src[i] - bm) * inv);
}

// ---- updates[b][n][d] = sum_m attn[b,n,m] * v[b,m,d] ----------------------
// One block per batch; 8 waves x 2 col-tiles = all 256 output cols, so the
// v stream (134 MB/iter) is staged into LDS exactly once, via async loads.
// grid = B, block = 256
__global__ void attn_v_kernel(const __bf16* __restrict__ attn_bf, // [B*16][4096]
                              const __bf16* __restrict__ kv,      // [B*4096][512] (v at +256)
                              float* __restrict__ updates) {      // [B*16][256]
  __shared__ __bf16 vtile[32 * 256];
  const int b    = blockIdx.x;
  const int wave = threadIdx.x >> 5;
  const int lane = threadIdx.x & 31;
  const int col0 = wave * 32;   // each wave: 2 adjacent 16-col tiles
  const __bf16* ab = attn_bf + (long)b * 16 * 4096;
  const __bf16* vb = kv + (long)b * 4096 * 512 + 256;

  const unsigned vtile_lds = (unsigned)(unsigned long long)(void*)vtile;
  const unsigned long long vbase = (unsigned long long)vb;

  v8f acc0 = {}, acc1 = {};
  for (int k0 = 0; k0 < 4096; k0 += 32) {
    __syncthreads();
    // async-stage 32 v-rows x 256 cols (16 KB) into LDS: 4 x B128 per thread
    for (int i = threadIdx.x; i < 1024; i += 256) {
      unsigned lds_off = vtile_lds + (unsigned)i * 16u;
      int r  = i >> 5;               // v-row within tile
      int cb = (i & 31) * 16;        // byte offset within row
      unsigned long long ga = vbase + (unsigned long long)(k0 + r) * 1024ull + (unsigned)cb;
      asm volatile("global_load_async_to_lds_b128 %0, %1, off"
                   :: "v"(lds_off), "v"(ga) : "memory");
    }
    asm volatile("s_wait_asynccnt 0x0" ::: "memory");
    if (k0 + 32 < 4096)
      __builtin_prefetch(vb + (long)(k0 + 32) * 512, 0, 1);
    __syncthreads();

    v16bf a = frag_bf16(ab + k0, 4096, lane);
    // B fragments: lane holds output column d, K values gathered down vtile
    const int off = (lane & 16) ? 8 : 0;
#pragma unroll
    for (int t = 0; t < 2; ++t) {
      int d = col0 + t * 16 + (lane & 15);
      const __bf16* p = vtile + (long)off * 256 + d;
      v16bf bm;
#pragma unroll
      for (int j = 0; j < 8; ++j) { bm[j] = p[j * 256]; bm[j + 8] = p[(16 + j) * 256]; }
      if (t == 0) acc0 = wmma_bf16(a, bm, acc0);
      else        acc1 = wmma_bf16(a, bm, acc1);
    }
  }

#pragma unroll
  for (int t = 0; t < 2; ++t) {
    const int d = col0 + t * 16 + (lane & 15);
    const v8f& acc = t == 0 ? acc0 : acc1;
#pragma unroll
    for (int i = 0; i < 8; ++i) {
      int n = i + ((lane & 16) ? 8 : 0);
      updates[((long)b * 16 + n) * 256 + d] = acc[i];
    }
  }
}

// ---- elementwise kernels --------------------------------------------------

__global__ void init_slots_kernel(const float* __restrict__ noise,
                                  const float* __restrict__ mu,
                                  const float* __restrict__ log_sigma,
                                  float* __restrict__ slots) {
  long i = (long)blockIdx.x * blockDim.x + threadIdx.x; // B*N*D = 262144
  int d = (int)(i & 255);
  slots[i] = mu[d] + expf(log_sigma[d]) * noise[i];
}

__device__ __forceinline__ float sigmoidf_(float x) { return 1.0f / (1.0f + expf(-x)); }

__global__ void gru_combine_kernel(const float* __restrict__ gx,
                                   const float* __restrict__ gh,
                                   float* __restrict__ slots) {
  long i = (long)blockIdx.x * blockDim.x + threadIdx.x; // 1024*256
  long row = i >> 8;
  int  d = (int)(i & 255);
  const float* gxr = gx + row * 768;
  const float* ghr = gh + row * 768;
  float r = sigmoidf_(gxr[d] + ghr[d]);
  float z = sigmoidf_(gxr[256 + d] + ghr[256 + d]);
  float n = tanhf(gxr[512 + d] + r * ghr[512 + d]);
  float prev = slots[i];
  slots[i] = (1.0f - z) * n + z * prev;
}

// one wave per row (8 rows per block of 256)
__global__ void layernorm_kernel(const float* __restrict__ slots,
                                 const float* __restrict__ g,
                                 const float* __restrict__ bta,
                                 float* __restrict__ out) {
  const int row  = blockIdx.x * 8 + (threadIdx.x >> 5);
  const int lane = threadIdx.x & 31;
  const float* x = slots + (long)row * 256;
  float s = 0.0f, s2 = 0.0f, v[8];
#pragma unroll
  for (int j = 0; j < 8; ++j) {
    float t = x[lane + j * 32];
    v[j] = t; s += t; s2 += t * t;
  }
#pragma unroll
  for (int o = 16; o > 0; o >>= 1) { s += __shfl_xor(s, o, 32); s2 += __shfl_xor(s2, o, 32); }
  const float mu  = s  * (1.0f / 256.0f);
  const float var = s2 * (1.0f / 256.0f) - mu * mu;
  const float inv = rsqrtf(var + LN_EPS_);
  float* o_ = out + (long)row * 256;
#pragma unroll
  for (int j = 0; j < 8; ++j) {
    int d = lane + j * 32;
    o_[d] = (v[j] - mu) * inv * g[d] + bta[d];
  }
}

__global__ void gelu_kernel(float* __restrict__ x, long n) {
  long i = (long)blockIdx.x * blockDim.x + threadIdx.x;
  if (i < n) {
    float t = x[i];
    x[i] = 0.5f * t * (1.0f + erff(t * 0.70710678118654752f));
  }
}

__global__ void residual_add_kernel(float* __restrict__ slots,
                                    const float* __restrict__ h2) {
  long i = (long)blockIdx.x * blockDim.x + threadIdx.x;
  slots[i] += h2[i];
}

// ---- weight transpose + f32 -> bf16 convert: Wt[n][k] = W[k][n] -----------
__global__ void transpose_bf16_kernel(const float* __restrict__ W,
                                      __bf16* __restrict__ Wt, int K, int N) {
  long i = (long)blockIdx.x * blockDim.x + threadIdx.x;
  if (i >= (long)K * N) return;
  long k = i / N;
  int  n = (int)(i % N);
  Wt[(long)n * K + k] = (__bf16)W[i];
}

// ---------------------------------------------------------------------------

extern "C" void kernel_launch(void* const* d_in, const int* in_sizes, int n_in,
                              void* d_out, int out_size, void* d_ws, size_t ws_size,
                              hipStream_t stream) {
  const float* inputs         = (const float*)d_in[0];
  const float* noise          = (const float*)d_in[1];
  const float* slot_mu        = (const float*)d_in[2];
  const float* slot_log_sigma = (const float*)d_in[3];
  const float* Wq   = (const float*)d_in[4];
  const float* Wk   = (const float*)d_in[5];
  const float* Wv   = (const float*)d_in[6];
  const float* W_ih = (const float*)d_in[7];
  const float* W_hh = (const float*)d_in[8];
  const float* b_ih = (const float*)d_in[9];
  const float* b_hh = (const float*)d_in[10];
  const float* ln_g = (const float*)d_in[11];
  const float* ln_b = (const float*)d_in[12];
  const float* W1   = (const float*)d_in[13];
  const float* b1   = (const float*)d_in[14];
  const float* W2   = (const float*)d_in[15];
  const float* b2   = (const float*)d_in[16];

  float* slots = (float*)d_out;  // slots live in d_out across iterations

  // workspace carve-out
  char* w = (char*)d_ws;
  auto alloc = [&](size_t bytes) {
    char* p = w;
    w += (bytes + 255) & ~(size_t)255;
    return p;
  };
  __bf16* kvT   = (__bf16*)alloc((size_t)512 * 256 * 2);      // [512][256] (WkT|WvT)
  __bf16* WqT   = (__bf16*)alloc((size_t)256 * 256 * 2);
  __bf16* WihT  = (__bf16*)alloc((size_t)768 * 256 * 2);
  __bf16* WhhT  = (__bf16*)alloc((size_t)768 * 256 * 2);
  __bf16* W1T   = (__bf16*)alloc((size_t)512 * 256 * 2);
  __bf16* W2T   = (__bf16*)alloc((size_t)256 * 512 * 2);
  __bf16* qbf   = (__bf16*)alloc((size_t)B_ * N_ * 256 * 2);          // [1024][256]
  __bf16* kv    = (__bf16*)alloc((size_t)B_ * M_ * 512 * 2);          // [B*M][512]
  float*  logits  = (float*)alloc((size_t)B_ * N_ * M_ * 4);          // [B][16][4096]
  __bf16* attn_bf = (__bf16*)alloc((size_t)B_ * N_ * M_ * 2);
  float*  updates = (float*)alloc((size_t)B_ * N_ * 256 * 4);
  float*  gx      = (float*)alloc((size_t)B_ * N_ * 768 * 4);
  float*  gh      = (float*)alloc((size_t)B_ * N_ * 768 * 4);
  float*  h_ln    = (float*)alloc((size_t)B_ * N_ * 256 * 4);
  float*  h1      = (float*)alloc((size_t)B_ * N_ * 512 * 4);
  float*  h2      = (float*)alloc((size_t)B_ * N_ * 256 * 4);
  (void)ws_size; (void)in_sizes; (void)n_in; (void)out_size;

  const int TB = 256;
  auto grid1 = [](long n) { return dim3((unsigned)((n + 255) / 256)); };

  // ---- weight prep (bf16, transposed to [out][in]) ----
  transpose_bf16_kernel<<<grid1(256 * 256), TB, 0, stream>>>(Wk, kvT, 256, 256);
  transpose_bf16_kernel<<<grid1(256 * 256), TB, 0, stream>>>(Wv, kvT + 256 * 256, 256, 256);
  transpose_bf16_kernel<<<grid1(256 * 256), TB, 0, stream>>>(Wq, WqT, 256, 256);
  transpose_bf16_kernel<<<grid1(256 * 768), TB, 0, stream>>>(W_ih, WihT, 256, 768);
  transpose_bf16_kernel<<<grid1(256 * 768), TB, 0, stream>>>(W_hh, WhhT, 256, 768);
  transpose_bf16_kernel<<<grid1(256 * 512), TB, 0, stream>>>(W1, W1T, 256, 512);
  transpose_bf16_kernel<<<grid1(512 * 256), TB, 0, stream>>>(W2, W2T, 512, 256);

  // ---- slot init ----
  init_slots_kernel<<<grid1((long)B_ * N_ * 256), TB, 0, stream>>>(
      noise, slot_mu, slot_log_sigma, slots);

  // ---- fused k/v projection: kv[B*M][512] = inputs @ [Wk|Wv]  (bf16 out) ----
  // CT=4 -> grid.y = 1: inputs streamed from HBM exactly once (via TDM).
  gemm_bf16t_kernel<4><<<dim3(B_ * M_ / 16, 1), TB, 0, stream>>>(
      inputs, 256, kvT, 256, nullptr, nullptr, kv, 512, 256);

  // ---- iterations ----
  for (int it = 0; it < ITERS_; ++it) {
    // q = slots @ Wq  -> bf16
    gemm_bf16t_kernel<2><<<dim3(B_ * N_ / 16, 1), TB, 0, stream>>>(
        slots, 256, WqT, 256, nullptr, nullptr, qbf, 256, 256);

    attn_logits_kernel<<<dim3(B_, M_ / 128), TB, 0, stream>>>(qbf, kv, logits);
    softmax_rows_kernel<<<dim3(B_ * N_), TB, 0, stream>>>(logits, attn_bf);
    attn_v_kernel<<<dim3(B_), TB, 0, stream>>>(attn_bf, kv, updates);

    // GRU gates
    gemm_bf16t_kernel<3><<<dim3(B_ * N_ / 16, 2), TB, 0, stream>>>(
        updates, 256, WihT, 256, b_ih, gx, nullptr, 768, 256);
    gemm_bf16t_kernel<3><<<dim3(B_ * N_ / 16, 2), TB, 0, stream>>>(
        slots, 256, WhhT, 256, b_hh, gh, nullptr, 768, 256);
    gru_combine_kernel<<<grid1((long)B_ * N_ * 256), TB, 0, stream>>>(gx, gh, slots);

    // LayerNorm + MLP residual
    layernorm_kernel<<<dim3(B_ * N_ / 8), TB, 0, stream>>>(slots, ln_g, ln_b, h_ln);
    gemm_bf16t_kernel<4><<<dim3(B_ * N_ / 16, 1), TB, 0, stream>>>(
        h_ln, 256, W1T, 256, b1, h1, nullptr, 512, 256);
    gelu_kernel<<<grid1((long)B_ * N_ * 512), TB, 0, stream>>>(h1, (long)B_ * N_ * 512);
    gemm_bf16t_kernel<2><<<dim3(B_ * N_ / 16, 1), TB, 0, stream>>>(
        h1, 512, W2T, 512, b2, h2, nullptr, 256, 512);
    residual_add_kernel<<<grid1((long)B_ * N_ * 256), TB, 0, stream>>>(slots, h2);
  }
}
